// PointNetSimple_61409442398998
// MI455X (gfx1250) — matrix operand order repaired and checked by reading, not compiled, not run on hardware
//
#include <hip/hip_runtime.h>

#define DEVN 16384
#define KNN  16
#define EPSF 1e-5f
#define SLICES 4
#define SRCS (DEVN / SLICES)   // sources per slice

typedef __attribute__((ext_vector_type(16))) _Float16 v16h;
typedef __attribute__((ext_vector_type(8)))  float    v8f;
typedef __attribute__((ext_vector_type(2)))  float    v2f;

#if __has_builtin(__builtin_amdgcn_wmma_f32_16x16x4_f32)
#define HAVE_WMMA_F32K4 1
#endif

// ---------------------------------------------------------------------------
// Kernel 1a: partial kNN over one source slice, WMMA-accelerated.
//   Rank sources by s_j = |p_j|^2 - 2*p_i.p_j (= d2 - |p_i|^2, same per-row
//   order). Source chunk staged in LDS as (-2x,-2y,-2z,|p|^2); one
//   V_WMMA_F32_16X16X4_F32 per 16x16 tile produces finished scores
//   (D = A(targets,xyz0) x B(-2*src) + C(init=|p_j|^2)). Scores go through a
//   padded LDS tile; each thread scans its own row. Source chunks are
//   double-buffer prefetched with GLOBAL_LOAD_ASYNC_TO_LDS.
//   Grid = 64 target-blocks x SLICES source-slices (occupancy: 2048 waves).
// ---------------------------------------------------------------------------
__global__ __launch_bounds__(256) void knn_partial_kernel(
    const float* __restrict__ pos, float* __restrict__ pd, int* __restrict__ pi) {
  extern __shared__ char kmem[];
  float4* stile = (float4*)kmem;                    // [64] (-2x,-2y,-2z,|p|^2)
  float*  dt    = (float*)(kmem + 1024);            // [256][68] padded scores
  float*  raw   = (float*)(kmem + 1024 + 256 * 68 * 4);  // [64*3] async landing

  const int tid  = threadIdx.x;
  const int lane = tid & 31;
  const int lo   = lane & 15, hi = lane >> 4;
  const int wv   = tid >> 5;
  const int tb    = blockIdx.x & 63;        // target block
  const int slice = blockIdx.x >> 6;        // source slice
  const int blockBase = tb * 256;
  const int s0 = slice * SRCS;

#ifdef HAVE_WMMA_F32K4
  // A fragments: this wave's two 16-target strips; lane holds row M=lo,
  // elements K = 2*hi + e  (K: x,y,z,0-pad)
  v2f afr[2];
#pragma unroll
  for (int s = 0; s < 2; ++s) {
    const float* tp = pos + 3 * (blockBase + (wv * 2 + s) * 16 + lo);
    afr[s][0] = (hi == 0) ? tp[0] : tp[2];
    afr[s][1] = (hi == 0) ? tp[1] : 0.f;
  }
#endif

  float bd[KNN];
  int   bi[KNN];
#pragma unroll
  for (int q = 0; q < KNN; ++q) { bd[q] = 3.4e38f; bi[q] = 0; }

  auto ins = [&](float d2, int j) {
    if (d2 < bd[KNN - 1]) {
      bd[KNN - 1] = d2; bi[KNN - 1] = j;
#pragma unroll
      for (int q = KNN - 1; q > 0; --q) {
        if (bd[q] < bd[q - 1]) {
          const float td = bd[q]; bd[q] = bd[q - 1]; bd[q - 1] = td;
          const int   ti = bi[q]; bi[q] = bi[q - 1]; bi[q - 1] = ti;
        }
      }
    }
  };

  auto issue_async = [&](int c0) {
    if (tid < 64) {
      const float* g = pos + 3 * (c0 + tid);
      const unsigned           ldst = (unsigned)(unsigned long long)&raw[3 * tid];
      const unsigned long long ga   = (unsigned long long)g;
      asm volatile("global_load_async_to_lds_b64 %0, %1, off\n\t"
                   "global_load_async_to_lds_b32 %0, %1, off offset:8"
                   :: "v"(ldst), "v"(ga) : "memory");
    }
  };

  issue_async(s0);  // prime the pipeline
  for (int c0 = s0; c0 < s0 + SRCS; c0 += 64) {
    asm volatile("s_wait_asynccnt 0x0" ::: "memory");
    __syncthreads();                    // raw[chunk c0] visible; dt free again
    if (tid < 64) {                     // transform raw -> (-2x,-2y,-2z,|p|^2)
      const float qx = raw[3 * tid + 0];
      const float qy = raw[3 * tid + 1];
      const float qz = raw[3 * tid + 2];
      float4 v;
      v.x = -2.f * qx; v.y = -2.f * qy; v.z = -2.f * qz;
      v.w = qx * qx + qy * qy + qz * qz;
      stile[tid] = v;
    }
    __syncthreads();                    // stile ready; raw free
    if (c0 + 64 < s0 + SRCS) issue_async(c0 + 64);  // overlap with compute+scan

    // --- score tiles: 2 strips x 4 col-tiles per wave ---
#pragma unroll
    for (int s = 0; s < 2; ++s) {
      const int trow0 = (wv * 2 + s) * 16;       // block-local target row base
#pragma unroll
      for (int ct = 0; ct < 4; ++ct) {
        const float4 sv = stile[ct * 16 + lo];
        v8f c;
#pragma unroll
        for (int r = 0; r < 8; ++r) c[r] = sv.w; // init with |p_j|^2
#ifdef HAVE_WMMA_F32K4
        v2f b;
        b[0] = (hi == 0) ? sv.x : sv.z;          // K = 2*hi + e  (-2*coord)
        b[1] = (hi == 0) ? sv.y : 0.f;           // K=3 pad
        c = __builtin_amdgcn_wmma_f32_16x16x4_f32(false, afr[s], false, b,
                                                  (short)0, c, false, false);
#else
#pragma unroll
        for (int r = 0; r < 8; ++r) {
          const float* tp = pos + 3 * (blockBase + trow0 + r + 8 * hi);
          c[r] += tp[0] * sv.x + tp[1] * sv.y + tp[2] * sv.z;
        }
#endif
#pragma unroll
        for (int r = 0; r < 8; ++r)
          dt[(trow0 + r + 8 * hi) * 68 + ct * 16 + lo] = c[r];
      }
    }
    __syncthreads();                    // dt ready

    // --- scan own row (NOT unrolled: keep the insert body small) ---
    const float* drow = dt + tid * 68;
#pragma unroll 1
    for (int u4 = 0; u4 < 16; ++u4) {
      const float4 dv = *(const float4*)(drow + u4 * 4);
      ins(dv.x, c0 + u4 * 4 + 0);
      ins(dv.y, c0 + u4 * 4 + 1);
      ins(dv.z, c0 + u4 * 4 + 2);
      ins(dv.w, c0 + u4 * 4 + 3);
    }
  }
  const long o = ((long)(blockBase + tid) * SLICES + slice) * KNN;
#pragma unroll
  for (int q = 0; q < KNN; ++q) { pd[o + q] = bd[q]; pi[o + q] = bi[q]; }
}

// ---------------------------------------------------------------------------
// Kernel 1b: merge the SLICES sorted partial top-16 lists per target.
// ---------------------------------------------------------------------------
__global__ void knn_merge_kernel(const float* __restrict__ pd,
                                 const int* __restrict__ pi,
                                 int* __restrict__ nbr) {
  const int i = blockIdx.x * blockDim.x + threadIdx.x;
  if (i >= DEVN) return;
  float bd[KNN];
  int   bi[KNN];
#pragma unroll
  for (int q = 0; q < KNN; ++q) { bd[q] = 3.4e38f; bi[q] = 0; }
  const float* p = pd + (long)i * SLICES * KNN;
  const int*   s = pi + (long)i * SLICES * KNN;
#pragma unroll 1
  for (int u = 0; u < SLICES * KNN; ++u) {
    const float d = p[u];
    if (d < bd[KNN - 1]) {
      bd[KNN - 1] = d; bi[KNN - 1] = s[u];
#pragma unroll
      for (int q = KNN - 1; q > 0; --q) {
        if (bd[q] < bd[q - 1]) {
          const float td = bd[q]; bd[q] = bd[q - 1]; bd[q - 1] = td;
          const int   ti = bi[q]; bi[q] = bi[q - 1]; bi[q - 1] = ti;
        }
      }
    }
  }
#pragma unroll
  for (int q = 0; q < KNN; ++q) nbr[i * KNN + q] = bi[q];
}

// ---------------------------------------------------------------------------
// Kernel 2: h0 = concat(pos, normal)  [N,6]
// ---------------------------------------------------------------------------
__global__ void concat_kernel(const float* __restrict__ pos,
                              const float* __restrict__ nor,
                              float* __restrict__ h0) {
  const int idx = blockIdx.x * blockDim.x + threadIdx.x;
  if (idx >= DEVN * 6) return;
  const int i = idx / 6, c = idx % 6;
  h0[idx] = (c < 3) ? pos[3 * i + c] : nor[3 * i + c - 3];
}

// ---------------------------------------------------------------------------
// Kernel 3: PointNet edge-conv, WMMA-based (unchanged).
// ---------------------------------------------------------------------------
template <int CIN, int KC1, int COUT>
__global__ __launch_bounds__(128) void conv_kernel(
    const float* __restrict__ x, const float* __restrict__ pos,
    const int* __restrict__ nbr,
    const float* __restrict__ Wa, const float* __restrict__ ba,
    const float* __restrict__ ga, const float* __restrict__ be,
    const float* __restrict__ Wb, const float* __restrict__ bb,
    float* __restrict__ out) {
  constexpr int KP1 = KC1 * 32;
  constexpr int KC2 = COUT / 32;
  constexpr int NT  = COUT / 16;
  static_assert(COUT % 32 == 0, "COUT must be multiple of 32");

  extern __shared__ char smem[];
  _Float16* lWa   = (_Float16*)smem;        // [KC1*NT][32 lanes][16] swizzled
  _Float16* lWb   = lWa + KP1 * COUT;       // [KC2*NT][32 lanes][16] swizzled
  _Float16* hmid0 = lWb + COUT * COUT;      // per-wave [KC2][32 lanes][16]
  float2* stats0  = (float2*)(hmid0 + (blockDim.x >> 5) * (16 * COUT));

  for (int idx = threadIdx.x; idx < KP1 * COUT; idx += blockDim.x) {
    const int t = idx & 15, l = (idx >> 4) & 31, f = idx >> 9;
    const int kc = f / NT, nt = f % NT;
    const int k = kc * 32 + ((l >> 4) << 3) + (t < 8 ? t : t + 8);
    const int n = nt * 16 + (l & 15);
    float v = 0.f;
    if (k < CIN + 3)       v = Wa[k * COUT + n];
    else if (k == CIN + 3) v = ba[n];                 // folded bias row
    lWa[idx] = (_Float16)v;
  }
  for (int idx = threadIdx.x; idx < COUT * COUT; idx += blockDim.x) {
    const int t = idx & 15, l = (idx >> 4) & 31, f = idx >> 9;
    const int kc = f / NT, nt = f % NT;
    const int k = kc * 32 + ((l >> 4) << 3) + (t < 8 ? t : t + 8);
    const int n = nt * 16 + (l & 15);
    lWb[idx] = (_Float16)Wb[k * COUT + n];
  }
  __syncthreads();

  const v16h* lWaV = (const v16h*)lWa;
  const v16h* lWbV = (const v16h*)lWb;

  const int lane = threadIdx.x & 31;
  const int lo = lane & 15, hi = lane >> 4;
  _Float16* hmid   = hmid0 + (threadIdx.x >> 5) * (16 * COUT);
  const v16h* hmidV = (const v16h*)hmid;
  float2* statsW   = stats0 + (threadIdx.x >> 5) * 256;   // [16 edges][16 groups]

  float bbv[NT];
#pragma unroll
  for (int nt = 0; nt < NT; ++nt) bbv[nt] = bb[nt * 16 + lo];

  v16h indf[KC2], gammaF[KC2], betaF[KC2];
#pragma unroll
  for (int kc = 0; kc < KC2; ++kc) {
#pragma unroll
    for (int t = 0; t < 16; ++t) {
      const int kb = kc * 32 + hi * 8 + (t < 8 ? t : t + 8);  // B-layout K
      indf[kc][t] = (_Float16)(((kb >> 3) == lo) ? 1.f : 0.f);
      const int ka = kc * 32 + hi * 8 + (t < 8 ? t : t + 8);  // A-layout K
      gammaF[kc][t] = (_Float16)ga[ka];
      betaF[kc][t]  = (_Float16)be[ka];
    }
  }

  const int wA = (((lo >> 3) & 1) * 16 + 8 * hi) * 16 + (lo & 7);

  const int wavesPerBlock = blockDim.x >> 5;
  const int wave   = blockIdx.x * wavesPerBlock + (threadIdx.x >> 5);
  const int nwaves = gridDim.x * wavesPerBlock;

  for (int i = wave; i < DEVN; i += nwaves) {
    const int j = nbr[i * KNN + lo];                 // edge = lo
    const float rx = pos[3 * j + 0] - pos[3 * i + 0];
    const float ry = pos[3 * j + 1] - pos[3 * i + 1];
    const float rz = pos[3 * j + 2] - pos[3 * i + 2];
    const float* __restrict__ xj = x + (long)j * CIN;

    v16h afr[KC1];
#pragma unroll
    for (int kc = 0; kc < KC1; ++kc) {
      if (kc * 32 + 31 < CIN) {
        const float4 f0 = *(const float4*)(xj + kc * 32 + hi * 8);
        const float4 f1 = *(const float4*)(xj + kc * 32 + hi * 8 + 4);
        const float4 f2 = *(const float4*)(xj + kc * 32 + hi * 8 + 16);
        const float4 f3 = *(const float4*)(xj + kc * 32 + hi * 8 + 20);
        afr[kc][0]  = (_Float16)f0.x; afr[kc][1]  = (_Float16)f0.y;
        afr[kc][2]  = (_Float16)f0.z; afr[kc][3]  = (_Float16)f0.w;
        afr[kc][4]  = (_Float16)f1.x; afr[kc][5]  = (_Float16)f1.y;
        afr[kc][6]  = (_Float16)f1.z; afr[kc][7]  = (_Float16)f1.w;
        afr[kc][8]  = (_Float16)f2.x; afr[kc][9]  = (_Float16)f2.y;
        afr[kc][10] = (_Float16)f2.z; afr[kc][11] = (_Float16)f2.w;
        afr[kc][12] = (_Float16)f3.x; afr[kc][13] = (_Float16)f3.y;
        afr[kc][14] = (_Float16)f3.z; afr[kc][15] = (_Float16)f3.w;
      } else {
#pragma unroll
        for (int t = 0; t < 16; ++t) {
          const int k = kc * 32 + hi * 8 + (t < 8 ? t : t + 8);
          float v;
          if (k < CIN)           v = xj[k];
          else if (k == CIN)     v = rx;
          else if (k == CIN + 1) v = ry;
          else if (k == CIN + 2) v = rz;
          else if (k == CIN + 3) v = 1.f;   // bias column
          else                   v = 0.f;
          afr[kc][t] = (_Float16)v;
        }
      }
    }

    // GEMM1: h = M @ Wa (+ba folded); scatter raw h into A-frag LDS layout
#pragma unroll
    for (int nt = 0; nt < NT; ++nt) {
      v8f c = {};
#pragma unroll
      for (int kc = 0; kc < KC1; ++kc) {
        const v16h b = lWaV[(kc * NT + nt) * 32 + lane];
        c = __builtin_amdgcn_wmma_f32_16x16x32_f16(false, afr[kc], false, b,
                                                   (short)0, c, false, false);
      }
#pragma unroll
      for (int r = 0; r < 8; ++r)
        hmid[wA + (nt >> 1) * 512 + r * 16 + (nt & 1) * 8] = (_Float16)c[r];
    }
    asm volatile("" ::: "memory");

    // group stats via WMMA against indicator matrix
    v16h ah[KC2];
#pragma unroll
    for (int kc = 0; kc < KC2; ++kc) ah[kc] = hmidV[kc * 32 + lane];

    v8f csum = {}, csq = {};
#pragma unroll
    for (int kc = 0; kc < KC2; ++kc) {
      const v16h ah2 = ah[kc] * ah[kc];               // v_pk_mul_f16
      csum = __builtin_amdgcn_wmma_f32_16x16x32_f16(false, ah[kc], false, indf[kc],
                                                    (short)0, csum, false, false);
      csq  = __builtin_amdgcn_wmma_f32_16x16x32_f16(false, ah2,    false, indf[kc],
                                                    (short)0, csq,  false, false);
    }
#pragma unroll
    for (int r = 0; r < 8; ++r) {
      const float mu  = csum[r] * 0.125f;
      const float ms  = csq[r] * 0.125f;
      const float var = fmaxf(ms - mu * mu, 0.f);
      float2 st; st.x = mu; st.y = rsqrtf(var + EPSF);
      statsW[(r + 8 * hi) * 16 + lo] = st;
    }
    asm volatile("" ::: "memory");

    // normalize + ReLU in registers; feed GEMM2 directly
    v16h a2[KC2];
#pragma unroll
    for (int kc = 0; kc < KC2; ++kc) {
      const float2 s0 = statsW[lo * 16 + kc * 4 + hi];
      const float2 s1 = statsW[lo * 16 + kc * 4 + hi + 2];
      const _Float16 m0 = (_Float16)s0.x, r0 = (_Float16)s0.y;
      const _Float16 m1 = (_Float16)s1.x, r1 = (_Float16)s1.y;
      v16h muv, rsv;
#pragma unroll
      for (int t = 0; t < 16; ++t) {
        muv[t] = (t < 8) ? m0 : m1;
        rsv[t] = (t < 8) ? r0 : r1;
      }
      v16h nv = (ah[kc] - muv) * rsv;
      nv = nv * gammaF[kc] + betaF[kc];
      const v16h z = {};
      a2[kc] = __builtin_elementwise_max(nv, z);
    }

    // GEMM2: h @ Wb, max over 16 edges, + bb, ReLU, store
#pragma unroll
    for (int nt = 0; nt < NT; ++nt) {
      v8f c = {};
#pragma unroll
      for (int kc = 0; kc < KC2; ++kc) {
        const v16h b = lWbV[(kc * NT + nt) * 32 + lane];
        c = __builtin_amdgcn_wmma_f32_16x16x32_f16(false, a2[kc], false, b,
                                                   (short)0, c, false, false);
      }
      float m = c[0];
#pragma unroll
      for (int r = 1; r < 8; ++r) m = fmaxf(m, c[r]);
      m = fmaxf(m, __int_as_float(__builtin_amdgcn_ds_swizzle(__float_as_int(m), 0x401f)));
      m = fmaxf(m + bbv[nt], 0.f);
      if (hi == 0) out[(long)i * COUT + nt * 16 + lo] = m;
    }
  }
}

// ---------------------------------------------------------------------------
extern "C" void kernel_launch(void* const* d_in, const int* in_sizes, int n_in,
                              void* d_out, int out_size, void* d_ws, size_t ws_size,
                              hipStream_t stream) {
  (void)in_sizes; (void)n_in; (void)out_size; (void)ws_size;

  const float* pos = (const float*)d_in[0];
  const float* nor = (const float*)d_in[1];
  const float* W1a = (const float*)d_in[2];  const float* b1a = (const float*)d_in[3];
  const float* g1  = (const float*)d_in[4];  const float* be1 = (const float*)d_in[5];
  const float* W1b = (const float*)d_in[6];  const float* b1b = (const float*)d_in[7];
  const float* W2a = (const float*)d_in[8];  const float* b2a = (const float*)d_in[9];
  const float* g2  = (const float*)d_in[10]; const float* be2 = (const float*)d_in[11];
  const float* W2b = (const float*)d_in[12]; const float* b2b = (const float*)d_in[13];
  const float* W3a = (const float*)d_in[14]; const float* b3a = (const float*)d_in[15];
  const float* g3  = (const float*)d_in[16]; const float* be3 = (const float*)d_in[17];
  const float* W3b = (const float*)d_in[18]; const float* b3b = (const float*)d_in[19];

  float* out = (float*)d_out;
  float* h1 = out;                          // [N,64]
  float* h2 = out + (size_t)DEVN * 64;      // [N,64]
  float* h3 = out + (size_t)DEVN * 128;     // [N,128]

  // workspace layout
  char* ws = (char*)d_ws;
  int*   nbr = (int*)ws;                                  ws += (size_t)DEVN * KNN * sizeof(int);
  float* h0  = (float*)ws;                                ws += (size_t)DEVN * 6 * sizeof(float);
  float* pd  = (float*)ws;                                ws += (size_t)DEVN * SLICES * KNN * sizeof(float);
  int*   pi  = (int*)ws;

  const size_t knnSh = 1024 + (size_t)256 * 68 * 4 + 64 * 3 * 4;
  knn_partial_kernel<<<64 * SLICES, 256, knnSh, stream>>>(pos, pd, pi);
  knn_merge_kernel<<<(DEVN + 255) / 256, 256, 0, stream>>>(pd, pi, nbr);
  concat_kernel<<<(DEVN * 6 + 255) / 256, 256, 0, stream>>>(pos, nor, h0);

  const int blocks = 1024;  // 4 waves/block -> 4096 waves, 4 nodes/wave
  const size_t statsSh = 4 * 256 * sizeof(float2);
  {
    size_t sh = (size_t)(1 * 32) * 64 * 2 + 64 * 64 * 2 + 4 * 16 * 64 * 2 + statsSh;
    conv_kernel<6, 1, 64><<<blocks, 128, sh, stream>>>(
        h0, pos, nbr, W1a, b1a, g1, be1, W1b, b1b, h1);
  }
  {
    size_t sh = (size_t)(3 * 32) * 64 * 2 + 64 * 64 * 2 + 4 * 16 * 64 * 2 + statsSh;
    conv_kernel<64, 3, 64><<<blocks, 128, sh, stream>>>(
        h1, pos, nbr, W2a, b2a, g2, be2, W2b, b2b, h2);
  }
  {
    size_t sh = (size_t)(3 * 32) * 128 * 2 + 128 * 128 * 2 + 4 * 16 * 128 * 2 + statsSh;
    conv_kernel<64, 3, 128><<<blocks, 128, sh, stream>>>(
        h2, pos, nbr, W3a, b3a, g3, be3, W3b, b3b, h3);
  }
}
